// IADGAT_7232724927267
// MI455X (gfx1250) — matrix operand order repaired and checked by reading, not compiled, not run on hardware
//
#include <hip/hip_runtime.h>

#define N_NODES 50000
#define F_IN    512
#define HID     32
#define HEADS   8
#define C1      (HEADS * HID)      // 256
#define N_CLS   32
#define NEDGE   800000
#define ETOT    (NEDGE + N_NODES)  // 850000 (edges + self loops)
#define NEG_SLOPE 0.2f

typedef __attribute__((ext_vector_type(16))) _Float16 v16h;
typedef __attribute__((ext_vector_type(8)))  float    v8f;

// ---------- monotone float <-> uint key (single-instruction atomic max) ----------
__device__ __forceinline__ unsigned ordKey(float f) {
    unsigned u = __float_as_uint(f);
    return (u & 0x80000000u) ? ~u : (u | 0x80000000u);
}
__device__ __forceinline__ float ordVal(unsigned k) {
    unsigned u = (k & 0x80000000u) ? (k & 0x7fffffffu) : ~k;
    return __uint_as_float(u);
}
__device__ __forceinline__ float leaky(float v) { return v > 0.f ? v : NEG_SLOPE * v; }

__device__ __forceinline__ void edge_sd(const int* __restrict__ ei, int e, int& src, int& dst) {
    if (e < NEDGE) { src = ei[e]; dst = ei[NEDGE + e]; }
    else           { src = dst = e - NEDGE; }            // self loops appended
}

// =====================================================================
// WMMA GEMM: D[M,N] = A[M,K] * B[K,N]   (f32 I/O, f16 multiply, f32 acc)
// Block: 256 threads (8 waves). Tile: 128 x BN, BK=32 (one WMMA K-step).
// Wave w computes rows [w*16, w*16+16) x BN (NT accumulators).
// Fragment layouts: A row-in-lane, B col-in-lane, both with contiguous k = 16*(lane>>4) + j.
// Out-of-range A rows are CLAMPED (not predicated): garbage rows only feed
// garbage D rows, which the store guard discards -> no exec-mask churn in
// the hot loop.
// =====================================================================
template <int BN>
__global__ __launch_bounds__(256) void gemm_wmma(const float* __restrict__ A,
                                                 const float* __restrict__ B,
                                                 float* __restrict__ D,
                                                 int M, int K, int N) {
    constexpr int BM = 128, BK = 32;
    constexpr int NT = BN / 16;
    __shared__ _Float16 lA[BM][BK];   // row-major rows
    __shared__ _Float16 lB[BN][BK];   // transposed: [n][k] so frag loads are contiguous

    const int tid  = threadIdx.x;
    const int wave = tid >> 5;
    const int lane = tid & 31;
    const int row0 = blockIdx.x * BM;
    const int n0   = blockIdx.y * BN;

    v8f acc[NT];
    for (int t = 0; t < NT; ++t)
        for (int i = 0; i < 8; ++i) acc[t][i] = 0.f;

    for (int k0 = 0; k0 < K; k0 += BK) {
        __syncthreads();
        // stage A tile (convert f32 -> f16); clamp row so address is always valid
        #pragma unroll
        for (int i = 0; i < (BM * BK / 4) / 256; ++i) {
            int idx = tid + i * 256;
            int r = idx >> 3, c4 = (idx & 7) << 2;        // BK/4 = 8 float4 per row
            int grow = row0 + r;
            grow = grow < M ? grow : (M - 1);
            float4 v = *(const float4*)(A + (size_t)grow * K + k0 + c4);
            lA[r][c4 + 0] = (_Float16)v.x; lA[r][c4 + 1] = (_Float16)v.y;
            lA[r][c4 + 2] = (_Float16)v.z; lA[r][c4 + 3] = (_Float16)v.w;
        }
        // stage B tile transposed
        #pragma unroll
        for (int i = 0; i < (BK * BN / 4) / 256; ++i) {
            int idx = tid + i * 256;
            int kr = idx / (BN / 4);
            int c4 = (idx % (BN / 4)) << 2;
            float4 v = *(const float4*)(B + (size_t)(k0 + kr) * N + n0 + c4);
            lB[c4 + 0][kr] = (_Float16)v.x; lB[c4 + 1][kr] = (_Float16)v.y;
            lB[c4 + 2][kr] = (_Float16)v.z; lB[c4 + 3][kr] = (_Float16)v.w;
        }
        __syncthreads();

        // A frag: lanes 0-15 row M=lane K=0..15; lanes 16-31 row M=lane-16 K=16..31
        const int arow = wave * 16 + (lane & 15);
        const int ka   = (lane >> 4) * 16;
        v16h a;
        #pragma unroll
        for (int j = 0; j < 16; ++j) a[j] = lA[arow][ka + j];

        #pragma unroll
        for (int t = 0; t < NT; ++t) {
            // B frag: lanes 0-15 col N=lane K=0..15; lanes 16-31 col N=lane-16 K=16..31
            const int bcol = t * 16 + (lane & 15);
            const int kb   = (lane >> 4) * 16;
            v16h b;
            #pragma unroll
            for (int j = 0; j < 16; ++j) b[j] = lB[bcol][kb + j];
            acc[t] = __builtin_amdgcn_wmma_f32_16x16x32_f16(
                false, a, false, b, (short)0, acc[t], false, false);
        }
    }

    // D layout: VGPR r -> M = r + (lane>>4)*8 ; N = lane&15
    const int mbase = row0 + wave * 16 + (lane >> 4) * 8;
    const int col   = lane & 15;
    #pragma unroll
    for (int t = 0; t < NT; ++t)
        #pragma unroll
        for (int r = 0; r < 8; ++r) {
            int grow = mbase + r;
            if (grow < M) D[(size_t)grow * N + n0 + t * 16 + col] = acc[t][r];
        }
}

// ---------- attention logits: al_s[n,h] = <h[n,h,:], a_src[h,:]> ----------
__global__ void attn_logits_k(const float* __restrict__ h, const float* __restrict__ a_s,
                              const float* __restrict__ a_d, float* __restrict__ al_s,
                              float* __restrict__ al_d, int n, int heads, int ch) {
    int i = blockIdx.x * blockDim.x + threadIdx.x;
    if (i >= n * heads) return;
    int node = i / heads, hd = i % heads;
    const float* hp = h + (size_t)node * heads * ch + hd * ch;
    const float* as = a_s + hd * ch;
    const float* ad = a_d + hd * ch;
    float ss = 0.f, sd = 0.f;
    for (int c = 0; c < ch; ++c) { float v = hp[c]; ss += v * as[c]; sd += v * ad[c]; }
    al_s[i] = ss; al_d[i] = sd;
}

// ---------- edge softmax pass 1: per-dst max via ordered-key atomicMax ----------
__global__ void edge_max_k(const int* __restrict__ ei, const float* __restrict__ al_s,
                           const float* __restrict__ al_d, unsigned* __restrict__ mkey,
                           int heads) {
    int e = blockIdx.x * blockDim.x + threadIdx.x;
    if (e >= ETOT) return;
    int src, dst; edge_sd(ei, e, src, dst);
    for (int h = 0; h < heads; ++h) {
        float v = leaky(al_s[src * heads + h] + al_d[dst * heads + h]);
        atomicMax(&mkey[dst * heads + h], ordKey(v));
    }
}

// ---------- edge softmax pass 2: per-dst sum of exp(e - m) ----------
__global__ void edge_sum_k(const int* __restrict__ ei, const float* __restrict__ al_s,
                           const float* __restrict__ al_d, const unsigned* __restrict__ mkey,
                           float* __restrict__ ssum, int heads) {
    int e = blockIdx.x * blockDim.x + threadIdx.x;
    if (e >= ETOT) return;
    int src, dst; edge_sd(ei, e, src, dst);
    for (int h = 0; h < heads; ++h) {
        float v = leaky(al_s[src * heads + h] + al_d[dst * heads + h]);
        float m = ordVal(mkey[dst * heads + h]);
        atomicAdd(&ssum[dst * heads + h], __expf(v - m));
    }
}

// ---------- layer-1 aggregation: one wave per edge, 8 f32 per lane ----------
__global__ __launch_bounds__(256) void agg1_k(const int* __restrict__ ei,
                                              const float* __restrict__ al_s,
                                              const float* __restrict__ al_d,
                                              const unsigned* __restrict__ mkey,
                                              const float* __restrict__ ssum,
                                              const float* __restrict__ h1,
                                              float* __restrict__ out) {
    int gw = (blockIdx.x * blockDim.x + threadIdx.x) >> 5;
    int lane = threadIdx.x & 31;
    if (gw >= ETOT) return;
    int src, dst; edge_sd(ei, gw, src, dst);
    int hd = lane >> 2;   // channels lane*8..lane*8+7 all belong to head lane/4
    float ev = leaky(al_s[src * HEADS + hd] + al_d[dst * HEADS + hd]);
    float m  = ordVal(mkey[dst * HEADS + hd]);
    float alpha = __expf(ev - m) / (ssum[dst * HEADS + hd] + 1e-16f);
    const float4* hs = (const float4*)(h1 + (size_t)src * C1 + lane * 8);
    float4 v0 = hs[0], v1 = hs[1];
    float* op = out + (size_t)dst * C1 + lane * 8;
    atomicAdd(op + 0, v0.x * alpha); atomicAdd(op + 1, v0.y * alpha);
    atomicAdd(op + 2, v0.z * alpha); atomicAdd(op + 3, v0.w * alpha);
    atomicAdd(op + 4, v1.x * alpha); atomicAdd(op + 5, v1.y * alpha);
    atomicAdd(op + 6, v1.z * alpha); atomicAdd(op + 7, v1.w * alpha);
}

// ---------- layer-2 aggregation: 8 threads per edge, 4 f32 each ----------
__global__ void agg2_k(const int* __restrict__ ei, const float* __restrict__ al_s,
                       const float* __restrict__ al_d, const unsigned* __restrict__ mkey,
                       const float* __restrict__ ssum, const float* __restrict__ h2,
                       float* __restrict__ out) {
    int t = blockIdx.x * blockDim.x + threadIdx.x;
    int e = t >> 3;
    if (e >= ETOT) return;
    int c0 = (t & 7) * 4;
    int src, dst; edge_sd(ei, e, src, dst);
    float ev = leaky(al_s[src] + al_d[dst]);
    float alpha = __expf(ev - ordVal(mkey[dst])) / (ssum[dst] + 1e-16f);
    float4 v = *(const float4*)(h2 + (size_t)src * N_CLS + c0);
    float* op = out + (size_t)dst * N_CLS + c0;
    atomicAdd(op + 0, v.x * alpha); atomicAdd(op + 1, v.y * alpha);
    atomicAdd(op + 2, v.z * alpha); atomicAdd(op + 3, v.w * alpha);
}

__global__ void elu_k(float* __restrict__ p, size_t n) {
    size_t i = (size_t)blockIdx.x * blockDim.x + threadIdx.x;
    if (i >= n) return;
    float x = p[i];
    p[i] = x > 0.f ? x : (__expf(x) - 1.f);
}

__global__ void deg_k(const int* __restrict__ ei, float* __restrict__ deg) {
    int e = blockIdx.x * blockDim.x + threadIdx.x;
    if (e >= ETOT) return;
    int src, dst; edge_sd(ei, e, src, dst);
    atomicAdd(&deg[dst], 1.f);
}

__global__ void dinv_k(float* __restrict__ deg, int n) {
    int i = blockIdx.x * blockDim.x + threadIdx.x;
    if (i >= n) return;
    float d = deg[i];
    deg[i] = d > 0.f ? rsqrtf(d) : 0.f;   // in place: deg -> deg^{-1/2}
}

__global__ void iconv_k(const int* __restrict__ ei, const float* __restrict__ dinv,
                        const float* __restrict__ h, float* __restrict__ out) {
    int t = blockIdx.x * blockDim.x + threadIdx.x;
    int e = t >> 3;
    if (e >= ETOT) return;
    int c0 = (t & 7) * 4;
    int src, dst; edge_sd(ei, e, src, dst);
    float nrm = dinv[src] * dinv[dst];
    float4 v = *(const float4*)(h + (size_t)src * N_CLS + c0);
    float* op = out + (size_t)dst * N_CLS + c0;
    atomicAdd(op + 0, v.x * nrm); atomicAdd(op + 1, v.y * nrm);
    atomicAdd(op + 2, v.z * nrm); atomicAdd(op + 3, v.w * nrm);
}

__global__ void zero_k(float* __restrict__ p, size_t n) {
    size_t i = (size_t)blockIdx.x * blockDim.x + threadIdx.x;
    if (i < n) p[i] = 0.f;
}
__global__ void bias_init_k(float* __restrict__ p, const float* __restrict__ b,
                            size_t n, int C) {
    size_t i = (size_t)blockIdx.x * blockDim.x + threadIdx.x;
    if (i < n) p[i] = b[i % C];
}

static inline int cdiv(long long a, long long b) { return (int)((a + b - 1) / b); }

extern "C" void kernel_launch(void* const* d_in, const int* in_sizes, int n_in,
                              void* d_out, int out_size, void* d_ws, size_t ws_size,
                              hipStream_t stream) {
    const float* x      = (const float*)d_in[0];
    const float* W1     = (const float*)d_in[1];
    const float* a_src1 = (const float*)d_in[2];
    const float* a_dst1 = (const float*)d_in[3];
    const float* b1     = (const float*)d_in[4];
    const float* W2     = (const float*)d_in[5];
    const float* a_src2 = (const float*)d_in[6];
    const float* a_dst2 = (const float*)d_in[7];
    const float* b2     = (const float*)d_in[8];
    const int*   ei     = (const int*)d_in[9];
    float* out = (float*)d_out;

    // workspace layout (re-initialized every call; ~123 MB)
    char* wp = (char*)d_ws;
    auto alloc = [&](size_t bytes) -> char* {
        char* p = wp; wp += (bytes + 255) & ~(size_t)255; return p;
    };
    float*    h1    = (float*)   alloc((size_t)N_NODES * C1 * 4);
    float*    out1  = (float*)   alloc((size_t)N_NODES * C1 * 4);
    float*    al_s1 = (float*)   alloc((size_t)N_NODES * HEADS * 4);
    float*    al_d1 = (float*)   alloc((size_t)N_NODES * HEADS * 4);
    unsigned* m1    = (unsigned*)alloc((size_t)N_NODES * HEADS * 4);
    float*    s1    = (float*)   alloc((size_t)N_NODES * HEADS * 4);
    float*    h2    = (float*)   alloc((size_t)N_NODES * N_CLS * 4);
    float*    out2  = (float*)   alloc((size_t)N_NODES * N_CLS * 4);
    float*    al_s2 = (float*)   alloc((size_t)N_NODES * 4);
    float*    al_d2 = (float*)   alloc((size_t)N_NODES * 4);
    unsigned* m2    = (unsigned*)alloc((size_t)N_NODES * 4);
    float*    s2    = (float*)   alloc((size_t)N_NODES * 4);
    float*    deg   = (float*)   alloc((size_t)N_NODES * 4);

    const int B = 256;

    // ---- init accumulators (every call: harness does not re-poison) ----
    zero_k<<<cdiv((long long)N_NODES * HEADS, B), B, 0, stream>>>((float*)m1, (size_t)N_NODES * HEADS);
    zero_k<<<cdiv((long long)N_NODES * HEADS, B), B, 0, stream>>>(s1, (size_t)N_NODES * HEADS);
    zero_k<<<cdiv(N_NODES, B), B, 0, stream>>>((float*)m2, N_NODES);
    zero_k<<<cdiv(N_NODES, B), B, 0, stream>>>(s2, N_NODES);
    zero_k<<<cdiv(N_NODES, B), B, 0, stream>>>(deg, N_NODES);
    zero_k<<<cdiv((long long)N_NODES * N_CLS, B), B, 0, stream>>>(out, (size_t)N_NODES * N_CLS);
    bias_init_k<<<cdiv((long long)N_NODES * C1, B), B, 0, stream>>>(out1, b1, (size_t)N_NODES * C1, C1);
    bias_init_k<<<cdiv((long long)N_NODES * N_CLS, B), B, 0, stream>>>(out2, b2, (size_t)N_NODES * N_CLS, N_CLS);

    // ---- layer 1: GEMM (WMMA), logits, softmax, aggregate, ELU ----
    gemm_wmma<64><<<dim3(cdiv(N_NODES, 128), C1 / 64), B, 0, stream>>>(x, W1, h1, N_NODES, F_IN, C1);
    attn_logits_k<<<cdiv((long long)N_NODES * HEADS, B), B, 0, stream>>>(h1, a_src1, a_dst1, al_s1, al_d1, N_NODES, HEADS, HID);
    edge_max_k<<<cdiv(ETOT, B), B, 0, stream>>>(ei, al_s1, al_d1, m1, HEADS);
    edge_sum_k<<<cdiv(ETOT, B), B, 0, stream>>>(ei, al_s1, al_d1, m1, s1, HEADS);
    agg1_k<<<cdiv((long long)ETOT * 32, B), B, 0, stream>>>(ei, al_s1, al_d1, m1, s1, h1, out1);
    elu_k<<<cdiv((long long)N_NODES * C1, B), B, 0, stream>>>(out1, (size_t)N_NODES * C1);

    // ---- layer 2 ----
    gemm_wmma<32><<<dim3(cdiv(N_NODES, 128), 1), B, 0, stream>>>(out1, W2, h2, N_NODES, C1, N_CLS);
    attn_logits_k<<<cdiv(N_NODES, B), B, 0, stream>>>(h2, a_src2, a_dst2, al_s2, al_d2, N_NODES, 1, N_CLS);
    edge_max_k<<<cdiv(ETOT, B), B, 0, stream>>>(ei, al_s2, al_d2, m2, 1);
    edge_sum_k<<<cdiv(ETOT, B), B, 0, stream>>>(ei, al_s2, al_d2, m2, s2, 1);
    agg2_k<<<cdiv((long long)ETOT * 8, B), B, 0, stream>>>(ei, al_s2, al_d2, m2, s2, h2, out2);

    // ---- GCN-normalized propagation ----
    deg_k<<<cdiv(ETOT, B), B, 0, stream>>>(ei, deg);
    dinv_k<<<cdiv(N_NODES, B), B, 0, stream>>>(deg, N_NODES);
    iconv_k<<<cdiv((long long)ETOT * 8, B), B, 0, stream>>>(ei, deg, out2, out);
}
